// Bilstm_8375186227367
// MI455X (gfx1250) — compile-verified, matrix operand order checked by reading, not run on hardware
//
#include <hip/hip_runtime.h>
#include <hip/hip_bf16.h>
#include <math.h>

#define B_   64
#define T_   256
#define E_   300
#define H_   150
#define NG   600      // 4*H
#define NT   38       // ceil(608/16) n-tiles (gate-interleaved, padded)
#define KT_E 10       // E padded to 320 -> 10 k-tiles of 32
#define KT_H 5        // H padded to 160 -> 5 k-tiles of 32
#define KT   15

typedef __attribute__((ext_vector_type(16))) __bf16 v16bf;
typedef __attribute__((ext_vector_type(8)))  float  v8f;

// workspace layout (bytes)
static constexpr size_t WB_BYTES = (size_t)2 * KT * NT * 512 * 2;        // 1,167,360
static constexpr size_t XA_BYTES = (size_t)2 * T_ * 4 * KT_E * 512 * 2;  // 20,971,520
static constexpr size_t HF_BYTES = (size_t)B_ * T_ * H_ * 4;             // 9,830,400

// ---- fragment index helpers (ISA 16-bit A/B layout, CDNA5 ISA 7.12.2) ----
// within a 16x32 (or 32x16) bf16 fragment stored as 512 contiguous bf16:
//   lane = (row&15) | (((kk>>3)&1)<<4),  e = (kk&7) | ((kk&16)>>1)
// (kk = k index within the 32-wide K tile; row = M for A, N for B)

// Pack weights into bf16 B-fragments, gate-interleaved columns: n' = 4*j + gate.
__global__ void pack_w(const float* __restrict__ wihf, const float* __restrict__ whhf,
                       const float* __restrict__ wihb, const float* __restrict__ whhb,
                       __bf16* __restrict__ WB) {
    int idx = blockIdx.x * blockDim.x + threadIdx.x;
    int total = 2 * KT * NT * 512;
    if (idx >= total) return;
    int e    = idx & 15;
    int lane = (idx >> 4) & 31;
    int frag = idx >> 9;
    int nt = frag % NT;
    int kt = (frag / NT) % KT;
    int d  = frag / (NT * KT);
    int kk = (e & 7) | ((lane >> 4) << 3) | ((e & 8) << 1);
    int np = nt * 16 + (lane & 15);            // gate-interleaved column
    float v = 0.f;
    if (np < NG) {
        int j = np >> 2, gate = np & 3;
        int no = gate * H_ + j;                // original row in (4H, *)
        const float* wih = d ? wihb : wihf;
        const float* whh = d ? whhb : whhf;
        if (kt < KT_E) { int k = kt * 32 + kk;          if (k < E_) v = wih[(size_t)no * E_ + k]; }
        else           { int k = (kt - KT_E) * 32 + kk; if (k < H_) v = whh[(size_t)no * H_ + k]; }
    }
    WB[(size_t)frag * 512 + lane * 16 + e] = (__bf16)v;
}

// Gather embedding + max-norm clip + pack into bf16 A-fragments.
// grid.x = B*T (step s), grid.y = direction; backward uses reversed index.
__global__ void pack_x(const int* __restrict__ x, const int* __restrict__ length,
                       const float* __restrict__ emb, __bf16* __restrict__ XA) {
    int tok = blockIdx.x;
    int d   = blockIdx.y;
    int b = tok / T_, s = tok % T_;
    int tid = threadIdx.x;                     // 64 threads
    int len = length[b];
    int tsrc = (d == 0) ? s : min(max(len - 1 - s, 0), T_ - 1);
    int row = x[b * T_ + tsrc];
    __shared__ float red[64];
    float ss = 0.f;
    for (int k = tid; k < E_; k += 64) { float v = emb[(size_t)row * E_ + k]; ss += v * v; }
    red[tid] = ss; __syncthreads();
    for (int off = 32; off > 0; off >>= 1) {
        if (tid < off) red[tid] += red[tid + off];
        __syncthreads();
    }
    float n = sqrtf(red[0]);
    float scale = fminf(1.f, 5.0f / fmaxf(n, 1e-7f));
    __bf16* base = XA + ((size_t)((d * T_ + s) * 4 + (b >> 4)) * KT_E) * 512;
    for (int k = tid; k < KT_E * 32; k += 64) {
        float v = (k < E_) ? emb[(size_t)row * E_ + k] * scale : 0.f;
        int kt = k >> 5, kk = k & 31;
        int lane = (b & 15) | (((kk >> 3) & 1) << 4);
        int e    = (kk & 7) | ((kk & 16) >> 1);
        base[(size_t)kt * 512 + lane * 16 + e] = (__bf16)v;
    }
}

// Persistent recurrent kernel: 8 blocks = (2 dirs x 4 batch tiles), 512 threads.
__global__ __launch_bounds__(512) void lstm(const __bf16* __restrict__ WB,
                                            const __bf16* __restrict__ XA,
                                            const float* __restrict__ bf,
                                            const float* __restrict__ bb,
                                            float* __restrict__ HF, float* __restrict__ HB) {
    const int d  = blockIdx.x >> 2;
    const int mt = blockIdx.x & 3;
    const int lane = threadIdx.x & 31;
    const int wave = threadIdx.x >> 5;

    __shared__ __align__(32) __bf16 hfrag[KT_H * 512];   // h in A-fragment layout
    for (int i = threadIdx.x; i < KT_H * 512; i += 512) hfrag[i] = (__bf16)0.f;

    const float* bias = d ? bb : bf;
    float* HOUT = d ? HB : HF;

    // Loop-carried *offset* (in bf16 elements) into WB, laundered per timestep.
    // Round 1: plain pointer -> LICM hoisted all 45 weight frags -> ~360 VGPR
    // scratch spills. Round 2: laundered integer pointer -> provenance lost ->
    // flat_load_b128 (ticks DScnt, fights the LDS pipe used by hfrag).
    // Now: base keeps global provenance (global_load_b128, LOADcnt only),
    // while the opaque offset still blocks hoisting.
    unsigned long long woff = (unsigned long long)d * KT * NT * 512;

    const v8f zero8 = {0.f, 0.f, 0.f, 0.f, 0.f, 0.f, 0.f, 0.f};
    v8f acc[3];
    v8f cst[3];
    float bvals[3][4];
    const int u = (lane & 15) >> 2;            // hidden-unit sub-index in tile
#pragma unroll
    for (int sl = 0; sl < 3; ++sl) {
        cst[sl] = zero8;
        int nt = wave + sl * 16;
        int j = nt * 4 + u;
#pragma unroll
        for (int g = 0; g < 4; ++g)
            bvals[sl][g] = (nt < NT && j < H_) ? bias[g * H_ + j] : 0.f;
    }
    __syncthreads();

    for (int t = 0; t < T_; ++t) {
        // Opaque per-iteration update of the weight offset (blocks LICM only).
        asm volatile("" : "+s"(woff));
        const __bf16* WBd = WB + woff;

        // --- g = [x_t | h] * W^T via WMMA; reads hfrag ---
#pragma unroll
        for (int sl = 0; sl < 3; ++sl) {
            int nt = wave + sl * 16;
            if (nt < NT) {
                acc[sl] = zero8;
                const __bf16* xa = XA + ((size_t)((d * T_ + t) * 4 + mt) * KT_E) * 512 + lane * 16;
#pragma unroll
                for (int kt = 0; kt < KT; ++kt) {
                    v16bf af;
                    if (kt < KT_E) af = *(const v16bf*)(xa + (size_t)kt * 512);
                    else           af = *(const v16bf*)(&hfrag[(kt - KT_E) * 512 + lane * 16]);
                    v16bf bfg = *(const v16bf*)(WBd + ((size_t)kt * NT + nt) * 512 + lane * 16);
                    acc[sl] = __builtin_amdgcn_wmma_f32_16x16x32_bf16(
                        false, af, false, bfg, (short)0, acc[sl], false, false);
                }
            }
        }
        __syncthreads();   // all reads of old hfrag done

        // --- gates, c/h update, write new hfrag + h to global ---
#pragma unroll
        for (int sl = 0; sl < 3; ++sl) {
            int nt = wave + sl * 16;
            if (nt < NT) {
                int j = nt * 4 + u;
                int src = (lane & 0x10) | (lane & 0x0C);   // base lane of this unit's 4 gates
#pragma unroll
                for (int r = 0; r < 8; ++r) {
                    float g0 = __shfl(acc[sl][r], src + 0, 32);
                    float g1 = __shfl(acc[sl][r], src + 1, 32);
                    float g2 = __shfl(acc[sl][r], src + 2, 32);
                    float g3 = __shfl(acc[sl][r], src + 3, 32);
                    float ig = 1.f / (1.f + __expf(-(g0 + bvals[sl][0])));
                    float fg = 1.f / (1.f + __expf(-(g1 + bvals[sl][1])));
                    float gg = tanhf(g2 + bvals[sl][2]);
                    float og = 1.f / (1.f + __expf(-(g3 + bvals[sl][3])));
                    float cn = fg * cst[sl][r] + ig * gg;
                    cst[sl][r] = cn;
                    float hv = og * tanhf(cn);
                    if ((lane & 3) == 0 && j < H_) {
                        int m  = r + ((lane >> 4) << 3);
                        int bg = mt * 16 + m;
                        HOUT[((size_t)bg * T_ + t) * H_ + j] = hv;
                        int kk = j & 31, kth = j >> 5;
                        int lf = m | (((kk >> 3) & 1) << 4);
                        int e  = (kk & 7) | ((kk & 16) >> 1);
                        hfrag[kth * 512 + lf * 16 + e] = (__bf16)hv;
                    }
                }
            }
        }
        __syncthreads();   // new hfrag visible before next step
    }
}

// Segment gather: left / tar / right, re-reversing h_b, mask -> 0.
__global__ void gather_out(const float* __restrict__ HF, const float* __restrict__ HB,
                           const int* __restrict__ local, const int* __restrict__ length,
                           const int* __restrict__ max_add, float* __restrict__ out, int out_size) {
    int o = blockIdx.x * blockDim.x + threadIdx.x;
    if (o >= out_size) return;
    int Lm = max_add[0], Tm = max_add[1], Rm = max_add[2];
    long n1 = (long)B_ * Lm * 300;
    long n2 = (long)B_ * Tm * 300;
    long idx = o; int seg, M;
    if (idx < n1)            { seg = 0; M = Lm; }
    else if (idx < n1 + n2)  { seg = 1; M = Tm; idx -= n1; }
    else                     { seg = 2; M = Rm; idx -= n1 + n2; }
    int c = (int)(idx % 300);
    long rem = idx / 300;
    int slot = (int)(rem % M);
    int b    = (int)(rem / M);
    int l0 = local[b * 2], l1 = local[b * 2 + 1], len = length[b];
    int start, seglen;
    if (seg == 0)      { start = 0;      seglen = l0; }
    else if (seg == 1) { start = l0;     seglen = l1 - l0 + 1; }
    else               { start = l1 + 1; seglen = len - l1 - 1; }
    float v = 0.f;
    if (slot < seglen) {
        int t = start + slot;
        v = (c < H_) ? HF[((size_t)b * T_ + t) * H_ + c]
                     : HB[((size_t)b * T_ + (len - 1 - t)) * H_ + (c - H_)];
    }
    out[o] = v;
}

extern "C" void kernel_launch(void* const* d_in, const int* in_sizes, int n_in,
                              void* d_out, int out_size, void* d_ws, size_t ws_size,
                              hipStream_t stream) {
    const int*   x      = (const int*)d_in[0];
    const int*   local  = (const int*)d_in[1];
    const int*   maxadd = (const int*)d_in[3];
    const int*   length = (const int*)d_in[4];
    const float* emb    = (const float*)d_in[5];
    const float* wihf   = (const float*)d_in[6];
    const float* whhf   = (const float*)d_in[7];
    const float* bf     = (const float*)d_in[8];
    const float* wihb   = (const float*)d_in[9];
    const float* whhb   = (const float*)d_in[10];
    const float* bb     = (const float*)d_in[11];
    float* out = (float*)d_out;

    char* ws = (char*)d_ws;
    __bf16* WB = (__bf16*)(ws);
    __bf16* XA = (__bf16*)(ws + WB_BYTES);
    float*  HF = (float*)(ws + WB_BYTES + XA_BYTES);
    float*  HB = (float*)(ws + WB_BYTES + XA_BYTES + HF_BYTES);

    int wtotal = 2 * KT * NT * 512;
    pack_w<<<(wtotal + 255) / 256, 256, 0, stream>>>(wihf, whhf, wihb, whhb, WB);
    pack_x<<<dim3(B_ * T_, 2), 64, 0, stream>>>(x, length, emb, XA);
    lstm<<<8, 512, 0, stream>>>(WB, XA, bf, bb, HF, HB);
    gather_out<<<(out_size + 255) / 256, 256, 0, stream>>>(HF, HB, local, length, maxadd, out, out_size);
}
